// LambdaResonatorNet_12996571038030
// MI455X (gfx1250) — matrix operand order repaired and checked by reading, not compiled, not run on hardware
//
#include <hip/hip_runtime.h>
#include <hip/hip_bf16.h>

typedef __attribute__((ext_vector_type(16))) _Float16 v16h;
typedef __attribute__((ext_vector_type(8)))  _Float16 v8h;
typedef __attribute__((ext_vector_type(8)))  float    v8f;

#define NB        4096
#define LATENT    256
#define INPUTD    784
#define NLAYERS   4
#define OUTD      10
#define DT_C      0.1f
#define PI_F      3.14159265358979f
#define TWO_PI_F  6.28318530717958f
#define INV2PI_F  0.159154943091895f
#define LDSS      264   // LDS row stride in halves: 132 dwords -> banks shift 4/row, conflict-free b128 reads

__device__ __forceinline__ v8f wmma16(v16h a, v16h b, v8f c) {
  return __builtin_amdgcn_wmma_f32_16x16x32_f16(false, a, false, b, (short)0, c,
                                                false, false);
}

__device__ __forceinline__ v16h cat8(v8h lo, v8h hi) {
  return __builtin_shufflevector(lo, hi, 0,1,2,3,4,5,6,7,8,9,10,11,12,13,14,15);
}

// ---- staging kernels: convert f32 weights to f16 in exact WMMA B-fragment layout ----
// B-fragment (32x16 f16, K x N): lane L holds N = L%16; halves h=0..15 hold
// K = (h<8 ? 8*(L/16) + h : 16 + 8*(L/16) + h-8).  Fragment stored lane-major: frag*512 + lane*16 + h halves.

// enc_w^T fragments: kt-major  dst[((kt*16+nt)*32+lane)*16+h], K padded 784->800 (25 ktiles)
__global__ __launch_bounds__(256) void stage_encw(const float* __restrict__ w,
                                                  _Float16* __restrict__ dst) {
  int idx = blockIdx.x * 256 + threadIdx.x;           // 25*16*32*16 = 204800
  if (idx >= 204800) return;
  int h    = idx & 15;
  int lane = (idx >> 4) & 31;
  int nt   = (idx >> 9) & 15;
  int kt   = idx >> 13;                               // 0..24
  int n = nt * 16 + (lane & 15);
  int k = kt * 32 + ((lane >> 4) * 8) + (h & 7) + ((h >> 3) * 16);
  float v = (k < INPUTD) ? w[n * INPUTD + k] : 0.0f;
  dst[idx] = (_Float16)v;
}

// kappa^T fragments: dst[(((layer*16+nt)*8+kt)*32+lane)*16+h];  B[k][n] = kappa[n][k]
__global__ __launch_bounds__(256) void stage_kappa(const float* __restrict__ kap,
                                                   _Float16* __restrict__ dst) {
  int idx = blockIdx.x * 256 + threadIdx.x;           // 4*16*8*32*16 = 262144
  if (idx >= 262144) return;
  int h     = idx & 15;
  int lane  = (idx >> 4) & 31;
  int kt    = (idx >> 9) & 7;
  int nt    = (idx >> 12) & 15;
  int layer = idx >> 16;
  int n = nt * 16 + (lane & 15);
  int k = kt * 32 + ((lane >> 4) * 8) + (h & 7) + ((h >> 3) * 16);
  dst[idx] = (_Float16)kap[((layer * LATENT) + n) * LATENT + k];
}

// dec_w^T fragments: dst[(kt*32+lane)*16+h];  N padded 10->16, K = 512
__global__ __launch_bounds__(256) void stage_decw(const float* __restrict__ w,
                                                  _Float16* __restrict__ dst) {
  int idx = blockIdx.x * 256 + threadIdx.x;           // 16*32*16 = 8192
  if (idx >= 8192) return;
  int h    = idx & 15;
  int lane = (idx >> 4) & 31;
  int kt   = idx >> 9;
  int n = lane & 15;
  int k = kt * 32 + ((lane >> 4) * 8) + (h & 7) + ((h >> 3) * 16);
  float v = (n < OUTD) ? w[n * (2 * LATENT) + k] : 0.0f;
  dst[idx] = (_Float16)v;
}

// ---- fused network: encoder GEMM + 4 resonator layers + decoder GEMM ----
// one wave (32 lanes) owns 16 batch rows; z kept as 16 f32 C-fragments in VGPRs.
__global__ __launch_bounds__(64) void resonator_fused(
    const float* __restrict__ x,        // [4096,784] f32
    const _Float16* __restrict__ encwB, // staged
    const _Float16* __restrict__ kappaB,// staged
    const _Float16* __restrict__ decwB, // staged
    const float* __restrict__ enc_b,    // [256]
    const float* __restrict__ omegas,   // [4,256]
    const float* __restrict__ dec_b,    // [10]
    float* __restrict__ out) {          // [4096,10]
  __shared__ __align__(16) _Float16 smem[2][2][16 * LDSS]; // [wave][sin/cos][16 x 264]

  const int lane = threadIdx.x & 31;
  const int wave = threadIdx.x >> 5;
  const int tile = blockIdx.x * 2 + wave;  // 0..255
  const int row0 = tile * 16;
  const int mrow = lane & 15;              // M (for A/C rows) or N (for B/C cols)
  const int hi   = lane >> 4;
  const int g    = hi ? 8 : 0;             // A-fragment K interleave offset

  _Float16* s_sin = &smem[wave][0][0];
  _Float16* s_cos = &smem[wave][1][0];

  v8f z[16];

  // ---------------- encoder: z = x @ enc_w^T + enc_b ----------------
#pragma unroll
  for (int nt = 0; nt < 16; ++nt) {
    float bias = enc_b[nt * 16 + mrow];
#pragma unroll
    for (int v = 0; v < 8; ++v) z[nt][v] = bias;
  }
  for (int kt = 0; kt < 25; ++kt) {
    // A 16x32 f16 fragment from raw f32 x (K tail padded with zeros; uniform branch)
    const float* xp = x + (size_t)(row0 + mrow) * INPUTD + kt * 32 + g;
    v8f x_lo = *(const v8f*)xp;
    v8f x_hi = {};
    if (kt < 24) x_hi = *(const v8f*)(xp + 16);
    v16h a;
#pragma unroll
    for (int i = 0; i < 8; ++i) {
      a[i]     = (_Float16)x_lo[i];
      a[i + 8] = (_Float16)x_hi[i];
    }
#pragma unroll
    for (int nt = 0; nt < 16; ++nt) {
      v16h b = *(const v16h*)(encwB + ((size_t)(kt * 16 + nt) * 32 + lane) * 16);
      z[nt] = wmma16(a, b, z[nt]);
    }
  }

  // ---------------- resonator layers ----------------
  for (int layer = 0; layer < NLAYERS; ++layer) {
    // write sin(z), cos(z) as f16 into per-wave LDS (C-layout -> [m][n] transpose)
#pragma unroll
    for (int nt = 0; nt < 16; ++nt) {
      int col = nt * 16 + mrow;
#pragma unroll
      for (int v = 0; v < 8; ++v) {
        float zz = z[nt][v];
        int m = v + 8 * hi;
        s_sin[m * LDSS + col] = (_Float16)__sinf(zz);
        s_cos[m * LDSS + col] = (_Float16)__cosf(zz);
      }
    }
    __syncthreads();

#pragma unroll
    for (int nt = 0; nt < 16; ++nt) {
      v8f accS = {};
      v8f accC = {};
      const _Float16* kb = kappaB + (size_t)((layer * 16 + nt) * 8) * 512;
#pragma unroll
      for (int kt = 0; kt < 8; ++kt) {
        const _Float16* rs = s_sin + mrow * LDSS + kt * 32;
        const _Float16* rc = s_cos + mrow * LDSS + kt * 32;
        v16h aS = cat8(*(const v8h*)(rs + g), *(const v8h*)(rs + 16 + g));
        v16h aC = cat8(*(const v8h*)(rc + g), *(const v8h*)(rc + 16 + g));
        v16h b  = *(const v16h*)(kb + ((size_t)kt * 32 + lane) * 16);
        accS = wmma16(aS, b, accS);   // (sin z) @ kappa^T
        accC = wmma16(aC, b, accC);   // (cos z) @ kappa^T
      }
      // epilogue: z += DT*(omega + sin*accC - cos*accS); wrap to [-pi, pi)
      float om = omegas[layer * LATENT + nt * 16 + mrow];
#pragma unroll
      for (int v = 0; v < 8; ++v) {
        float zz  = z[nt][v];
        float s   = __sinf(zz);
        float c   = __cosf(zz);
        float upd = s * accC[v] - c * accS[v];
        float t   = zz + DT_C * (om + upd) + PI_F;
        t -= TWO_PI_F * floorf(t * INV2PI_F);
        z[nt][v] = t - PI_F;
      }
    }
    __syncthreads();
  }

  // ---------------- decoder: [sin z | cos z] @ dec_w^T + dec_b ----------------
#pragma unroll
  for (int nt = 0; nt < 16; ++nt) {
    int col = nt * 16 + mrow;
#pragma unroll
    for (int v = 0; v < 8; ++v) {
      float zz = z[nt][v];
      int m = v + 8 * hi;
      s_sin[m * LDSS + col] = (_Float16)__sinf(zz);
      s_cos[m * LDSS + col] = (_Float16)__cosf(zz);
    }
  }
  __syncthreads();

  float db = (mrow < OUTD) ? dec_b[mrow] : 0.0f;
  v8f acc;
#pragma unroll
  for (int v = 0; v < 8; ++v) acc[v] = db;
#pragma unroll
  for (int kt = 0; kt < 16; ++kt) {
    const _Float16* buf = (kt < 8) ? s_sin : s_cos;
    int kbase = (kt & 7) * 32;
    const _Float16* r = buf + mrow * LDSS + kbase;
    v16h a = cat8(*(const v8h*)(r + g), *(const v8h*)(r + 16 + g));
    v16h b = *(const v16h*)(decwB + ((size_t)kt * 32 + lane) * 16);
    acc = wmma16(a, b, acc);
  }
  if (mrow < OUTD) {
#pragma unroll
    for (int v = 0; v < 8; ++v) {
      int m = v + 8 * hi;
      out[(size_t)(row0 + m) * OUTD + mrow] = acc[v];
    }
  }
}

extern "C" void kernel_launch(void* const* d_in, const int* in_sizes, int n_in,
                              void* d_out, int out_size, void* d_ws, size_t ws_size,
                              hipStream_t stream) {
  (void)in_sizes; (void)n_in; (void)out_size; (void)ws_size;
  const float* x      = (const float*)d_in[0];
  const float* enc_w  = (const float*)d_in[1];
  const float* enc_b  = (const float*)d_in[2];
  const float* kappas = (const float*)d_in[3];
  const float* omegas = (const float*)d_in[4];
  const float* dec_w  = (const float*)d_in[5];
  const float* dec_b  = (const float*)d_in[6];
  float* out = (float*)d_out;

  char* ws = (char*)d_ws;
  _Float16* encwB  = (_Float16*)(ws);                 // 204800 halves = 409600 B
  _Float16* kappaB = (_Float16*)(ws + 409600);        // 262144 halves = 524288 B
  _Float16* decwB  = (_Float16*)(ws + 409600 + 524288); // 8192 halves = 16384 B

  stage_encw <<<(204800 + 255) / 256, 256, 0, stream>>>(enc_w, encwB);
  stage_kappa<<<262144 / 256,         256, 0, stream>>>(kappas, kappaB);
  stage_decw <<<8192 / 256,           256, 0, stream>>>(dec_w, decwB);

  // 4096 rows / (16 rows per wave * 2 waves per block) = 128 blocks
  resonator_fused<<<128, 64, 0, stream>>>(x, encwB, kappaB, decwB,
                                          enc_b, omegas, dec_b, out);
}
